// StackedClockworkRNN_31834297598566
// MI455X (gfx1250) — matrix operand order repaired and checked by reading, not compiled, not run on hardware
//
#include <hip/hip_runtime.h>

typedef __attribute__((ext_vector_type(16))) __bf16   v16bf;
typedef __attribute__((ext_vector_type(8)))  float    v8f;
typedef __attribute__((ext_vector_type(4)))  unsigned u32x4;

namespace {
constexpr int kB = 32, kT = 512, kH = 1024, kD = 4;
constexpr int kNB = 16;   // persistent blocks
// workspace layout (bytes)
constexpr size_t OFF_BAR = 0;                                   // 2 uints: counter, gen
constexpr size_t OFF_WXS = 256;                                 // bf16 swizzled Wx  [D][64][32][32][16]
constexpr size_t SZ_W    = (size_t)kD * 64 * 32 * 512 * 2;      // 8 MiB
constexpr size_t OFF_WHS = OFF_WXS + SZ_W;                      // bf16 swizzled Wh (masked)
constexpr size_t OFF_HST = OFF_WHS + SZ_W;                      // f32  h-state [D][B][H]
constexpr size_t SZ_HST  = (size_t)kD * kB * kH * 4;
constexpr size_t OFF_HBF = OFF_HST + SZ_HST;                    // bf16 h double-buffer [2][D][B][H]
constexpr size_t SZ_HBF  = (size_t)2 * kD * kB * kH * 2;
constexpr size_t OFF_XBF = OFF_HBF + SZ_HBF;                    // bf16 x, reordered [T][B][H]
}

__device__ __forceinline__ unsigned short f2bf(float f) {
    union { float f; unsigned u; } a; a.f = f;
    unsigned r = a.u + 0x7FFFu + ((a.u >> 16) & 1u);            // RNE
    return (unsigned short)(r >> 16);
}

// ---------------- prologue kernels ----------------

__global__ void init_kernel(unsigned char* ws) {
    size_t i = (size_t)blockIdx.x * blockDim.x + threadIdx.x;
    if (i < 2) ((unsigned*)(ws + OFF_BAR))[i] = 0u;
    if (i < (size_t)kD * kB * kH) ((float*)(ws + OFF_HST))[i] = 0.0f;
    if (i < (size_t)2 * kD * kB * kH) ((unsigned short*)(ws + OFF_HBF))[i] = 0;
}

__global__ void xconvert_kernel(const float* __restrict__ x, unsigned char* ws) {
    // out idx enumerates [t][b][c]; input is [b][t][c]
    size_t idx = (size_t)blockIdx.x * blockDim.x + threadIdx.x;   // 0 .. 16777215
    unsigned c = (unsigned)(idx & 1023u);
    unsigned b = (unsigned)((idx >> 10) & 31u);
    unsigned t = (unsigned)(idx >> 15);
    ((unsigned short*)(ws + OFF_XBF))[idx] =
        f2bf(x[(((size_t)b * kT + t) << 10) + c]);
}

__global__ void wswizzle_kernel(const float* __restrict__ Wx0,
                                const float* __restrict__ Wxd,
                                const float* __restrict__ Wh,
                                const int*   __restrict__ periods,
                                unsigned char* ws) {
    size_t idx = (size_t)blockIdx.x * blockDim.x + threadIdx.x;   // 0 .. 2*2^22-1
    unsigned tensor = (unsigned)(idx >> 22);
    unsigned rem    = (unsigned)(idx & 0x3FFFFFu);
    unsigned d  = (rem >> 20) & 3u;
    unsigned nt = (rem >> 14) & 63u;
    unsigned kt = (rem >> 9)  & 31u;
    unsigned ln = (rem >> 4)  & 31u;
    unsigned q  =  rem        & 15u;
    // B-fragment layout: lane<16 -> col = nt*16+lane, K = kt*32 + 0..15
    //                    lane>=16 -> same col, K = kt*32 + 16..31
    unsigned k = kt * 32u + (ln >> 4) * 16u + q;
    unsigned n = nt * 16u + (ln & 15u);
    if (tensor == 0) {
        float v = (d == 0) ? Wx0[(size_t)k * kH + n]
                           : Wxd[(((size_t)(d - 1) * kH) + k) * kH + n];
        ((unsigned short*)(ws + OFF_WXS))[rem] = f2bf(v);
    } else {
        float v = Wh[(((size_t)d * kH) + k) * kH + n];
        if (periods[k] < periods[n]) v = 0.0f;                    // clockwork mask
        ((unsigned short*)(ws + OFF_WHS))[rem] = f2bf(v);
    }
}

// ---------------- persistent main kernel ----------------

__device__ __forceinline__ void grid_sync(unsigned* bar, unsigned expect, unsigned nblk) {
    __threadfence();
    __syncthreads();
    if (threadIdx.x == 0) {
        unsigned prev = __hip_atomic_fetch_add(&bar[0], 1u, __ATOMIC_ACQ_REL,
                                               __HIP_MEMORY_SCOPE_AGENT);
        if (prev == nblk - 1u) {
            __hip_atomic_store(&bar[0], 0u, __ATOMIC_RELAXED, __HIP_MEMORY_SCOPE_AGENT);
            __hip_atomic_store(&bar[1], expect, __ATOMIC_RELEASE, __HIP_MEMORY_SCOPE_AGENT);
        } else {
            // hot poll: barrier wake-up latency is the critical path of the RNN
            while (__hip_atomic_load(&bar[1], __ATOMIC_ACQUIRE,
                                     __HIP_MEMORY_SCOPE_AGENT) != expect) { }
        }
    }
    __syncthreads();
    __threadfence();
}

__global__ __launch_bounds__(256)
void cwrnn_persistent(const float* __restrict__ bias,     // [D][H]
                      const int*   __restrict__ periods,  // [H]
                      float*       __restrict__ out,      // [B][T][D][H]
                      unsigned char* ws) {
    unsigned*             bar  = (unsigned*)(ws + OFF_BAR);
    const unsigned short* Wxs  = (const unsigned short*)(ws + OFF_WXS);
    const unsigned short* Whs  = (const unsigned short*)(ws + OFF_WHS);
    float*                hst  = (float*)(ws + OFF_HST);
    unsigned short*       hbf  = (unsigned short*)(ws + OFF_HBF);
    const unsigned short* xbf  = (const unsigned short*)(ws + OFF_XBF);

    const int lane  = threadIdx.x & 31;
    const int gw    = blockIdx.x * 8 + (threadIdx.x >> 5);  // 128 waves
    const int m0    = (gw & 1) * 16;                        // M tile (batch rows)
    const int ntile = gw >> 1;                              // 0..63
    const int n0    = ntile * 16;
    const int rhi   = lane >> 4;                            // half-wave
    const int arow  = m0 + (lane & 15);                     // A row for this lane
    const int ncol  = n0 + (lane & 15);                     // output column

    // wave-uniform scalars: force into SGPRs so all branches/trip-counts are
    // scalar and EXEC stays all-ones around every WMMA (ISA requirement).
    const int pn = __builtin_amdgcn_readfirstlane(periods[n0]);   // pow2 period
    int kh = 0;
    for (int kt = 0; kt < 32; ++kt)
        if (periods[kt * 32 + 31] < pn) kh = kt + 1;
    const int khstart = __builtin_amdgcn_readfirstlane(kh);       // multiple of 8

    // per-depth bias for this lane's column, hoisted out of the time loop
    float bb4[kD];
    #pragma unroll
    for (int d = 0; d < kD; ++d) bb4[d] = bias[d * kH + ncol];

    // A fragment: contiguous K order matching B (lane<16: K 0..15; lane>=16: K 16..31)
    auto loadA = [&](const unsigned short* src, int kt) -> v16bf {
        const unsigned short* p = src + (size_t)arow * kH + kt * 32 + rhi * 16;
        union { v16bf v; u32x4 u[2]; } ua;
        ua.u[0] = *(const u32x4*)(p);
        ua.u[1] = *(const u32x4*)(p + 8);
        return ua.v;
    };
    // B fragment: pre-swizzled, 16 halfs per lane, contiguous
    auto loadB = [&](const unsigned short* wb, int d, int kt) -> v16bf {
        const unsigned short* p = wb + ((((size_t)d * 64 + ntile) * 32 + kt) << 9) + lane * 16;
        union { v16bf v; u32x4 u[2]; } ub;
        ub.u[0] = *(const u32x4*)(p);
        ub.u[1] = *(const u32x4*)(p + 8);
        return ub.v;
    };

    unsigned bexp = 0;
    for (int t = 1; t <= kT; ++t) {
        const int curp = t & 1;
        unsigned short* hb_cur = hbf + (size_t)curp * (kD * kB * kH);
        const unsigned short* hb_prev = hbf + (size_t)(curp ^ 1) * (kD * kB * kH);
        const bool active = ((t & (pn - 1)) == 0);          // scalar (pn in SGPR)

        for (int d = 0; d < kD; ++d) {
            float hd[8];
            if (active) {
                v8f acc = {0.f, 0.f, 0.f, 0.f, 0.f, 0.f, 0.f, 0.f};
                const unsigned short* Asrc = (d == 0)
                    ? (xbf + (size_t)(t - 1) * kB * kH)
                    : (hb_cur + (size_t)(d - 1) * kB * kH); // written before last barrier
                for (int kt = 0; kt < 32; kt += 4) {        // exact multiple, no remainder
                    #pragma unroll
                    for (int j = 0; j < 4; ++j) {
                        v16bf a = loadA(Asrc, kt + j);
                        v16bf b = loadB(Wxs, d, kt + j);
                        acc = __builtin_amdgcn_wmma_f32_16x16x32_bf16(
                                  false, a, false, b, (short)0, acc, false, false);
                    }
                }
                const unsigned short* Hsrc = hb_prev + (size_t)d * kB * kH;
                for (int kt = khstart; kt < 32; kt += 4) {  // khstart % 8 == 0
                    #pragma unroll
                    for (int j = 0; j < 4; ++j) {
                        v16bf a = loadA(Hsrc, kt + j);
                        v16bf b = loadB(Whs, d, kt + j);
                        acc = __builtin_amdgcn_wmma_f32_16x16x32_bf16(
                                  false, a, false, b, (short)0, acc, false, false);
                    }
                }
                #pragma unroll
                for (int r = 0; r < 8; ++r) hd[r] = tanhf(acc[r] + bb4[d]);
            } else {
                #pragma unroll
                for (int r = 0; r < 8; ++r)
                    hd[r] = hst[((size_t)d * kB + (m0 + r + 8 * rhi)) * kH + ncol];
            }
            // C/D layout: VGPR r, lanes0-15 -> M=r, lanes16-31 -> M=r+8
            #pragma unroll
            for (int r = 0; r < 8; ++r) {
                const int bi = m0 + r + 8 * rhi;
                out[(((size_t)bi * kT + (t - 1)) * kD + d) * kH + ncol] = hd[r];
                hb_cur[((size_t)d * kB + bi) * kH + ncol] = f2bf(hd[r]);
                if (active)
                    hst[((size_t)d * kB + bi) * kH + ncol] = hd[r];
            }
            // Barriers only where a cross-wave edge exists: after d0/d1/d2.
            // t.d3 -> (t+1).* edges are ordered by the (t+1).d0/d1/d2 barriers,
            // since each block fences its d3 stores before arriving at (t+1).d0.
            if (d != kD - 1) {
                ++bexp;
                grid_sync(bar, bexp, (unsigned)gridDim.x);
            }
        }
    }
}

// ---------------- host entry ----------------

extern "C" void kernel_launch(void* const* d_in, const int* in_sizes, int n_in,
                              void* d_out, int out_size, void* d_ws, size_t ws_size,
                              hipStream_t stream) {
    const float* x       = (const float*)d_in[0];
    const float* Wx0     = (const float*)d_in[1];
    const float* Wxd     = (const float*)d_in[2];
    const float* Wh      = (const float*)d_in[3];
    const float* bias    = (const float*)d_in[4];
    const int*   periods = (const int*)d_in[5];
    unsigned char* ws    = (unsigned char*)d_ws;
    float* out           = (float*)d_out;
    (void)in_sizes; (void)n_in; (void)out_size; (void)ws_size;

    init_kernel<<<1024, 256, 0, stream>>>(ws);                       // 262144 threads
    xconvert_kernel<<<65536, 256, 0, stream>>>(x, ws);               // 16.7M elems
    wswizzle_kernel<<<32768, 256, 0, stream>>>(Wx0, Wxd, Wh, periods, ws); // 8.4M elems
    cwrnn_persistent<<<kNB, 256, 0, stream>>>(bias, periods, out, ws);
}
